// MultiSAGENet_86088324481901
// MI455X (gfx1250) — matrix operand order repaired and checked by reading, not compiled, not run on hardware
//
#include <hip/hip_runtime.h>
#include <math.h>

// ---------------------------------------------------------------------------
// CDNA5 (gfx1250) MultiSAGE pipeline.
// All matmuls via v_wmma_f32_16x16x32_bf16 (bf16 A/B, f32 accumulate).
// Activations that feed the edge gather are kept in bf16 to halve the
// dominant HBM traffic (1.6M edges x 64 feats x 3 layers).
// Segment sum via global_atomic_add_f32 (unsafeAtomicAdd), segment max via
// monotonic-uint encoding + global_atomic_max_u32.
// ---------------------------------------------------------------------------

typedef __attribute__((ext_vector_type(16))) __bf16 v16bf;
typedef __attribute__((ext_vector_type(8)))  float  v8f;

#define ENC_NEGINF 0x007FFFFFu   // encOrd(-inf)

__device__ __forceinline__ unsigned short f2bf(float f) {
  unsigned u = __float_as_uint(f);
  unsigned r = ((u >> 16) & 1u) + 0x7FFFu;   // round to nearest even
  return (unsigned short)((u + r) >> 16);
}
__device__ __forceinline__ float bf2f(unsigned short h) {
  return __uint_as_float(((unsigned)h) << 16);
}
__device__ __forceinline__ unsigned pack2(float a, float b) {
  return (unsigned)f2bf(a) | ((unsigned)f2bf(b) << 16);
}
// monotonic float<->uint for atomicMax on floats
__device__ __forceinline__ unsigned encOrd(float f) {
  unsigned u = __float_as_uint(f);
  return (u & 0x80000000u) ? ~u : (u | 0x80000000u);
}
__device__ __forceinline__ float decOrd(unsigned u) {
  return (u & 0x80000000u) ? __uint_as_float(u & 0x7FFFFFFFu) : __uint_as_float(~u);
}
__device__ __forceinline__ float siluf(float v) { return v / (1.0f + __expf(-v)); }
__device__ __forceinline__ void atomAddF(float* p, float v) { unsafeAtomicAdd(p, v); }

union FragU { unsigned u[8]; v16bf v; };

__device__ __forceinline__ v8f wmma_bf16x8(const unsigned a[8], const unsigned b[8], v8f c) {
  FragU fa, fb;
#pragma unroll
  for (int i = 0; i < 8; ++i) { fa.u[i] = a[i]; fb.u[i] = b[i]; }
  return __builtin_amdgcn_wmma_f32_16x16x32_bf16(false, fa.v, false, fb.v, (short)0, c, false, false);
}

// A fragment: 16x32 bf16, row-major source with stride lda (elements).
// lane 0-15: M=lane, K-pairs {0..7,16..23}; lane 16-31: M=lane-16 (per ISA:
// same M rows, K offset +8 / +24). Each VGPR j holds K pair starting at
// k0 = 2j + 8*hi + (j>=4 ? 8 : 0)  -> contiguous, one dword load.
__device__ __forceinline__ void load_fragA(const unsigned short* __restrict__ A, int lda,
                                           int row0, int maxRow, int kOff, int lane,
                                           unsigned au[8]) {
  int r = row0 + (lane & 15);
  if (r > maxRow) r = maxRow;                 // clamp: keep EXEC all-ones for WMMA
  int hi = (lane >> 4) & 1;
  const unsigned short* p = A + (size_t)r * lda + kOff;
#pragma unroll
  for (int j = 0; j < 8; ++j) {
    int k0 = 2 * j + 8 * hi + ((j >= 4) ? 8 : 0);
    au[j] = *(const unsigned*)(p + k0);
  }
}
// B fragment: 32x16 bf16, weights stored TRANSPOSED (N x K row-major, stride
// ldb); K pairs match A: k0 = 2j + 8*hi + (j>=4 ? 8 : 0).
__device__ __forceinline__ void load_fragB(const unsigned short* __restrict__ Wt, int ldb,
                                           int col0, int kOff, int lane, unsigned bu[8]) {
  int c = col0 + (lane & 15);
  int hi = (lane >> 4) & 1;
  const unsigned short* p = Wt + (size_t)c * ldb + kOff;
#pragma unroll
  for (int j = 0; j < 8; ++j) bu[j] = *(const unsigned*)(p + 2 * j + 8 * hi + ((j >= 4) ? 8 : 0));
}

// ------------------------- utility kernels ---------------------------------
__global__ void fill_u32_kernel(unsigned* p, unsigned v, long long n) {
  long long t = blockIdx.x * (long long)blockDim.x + threadIdx.x;
  if (t < n) p[t] = v;
}
__global__ void fill_f32_kernel(float* p, float v, long long n) {
  long long t = blockIdx.x * (long long)blockDim.x + threadIdx.x;
  if (t < n) p[t] = v;
}
__global__ void cast_f32_bf16_kernel(const float* __restrict__ s, unsigned short* __restrict__ d,
                                     long long n) {
  long long t = blockIdx.x * (long long)blockDim.x + threadIdx.x;
  if (t < n) d[t] = f2bf(s[t]);
}
// src: K x N row-major f32  ->  dst[n*ldk + kOff + k] bf16 (transpose+cast)
__global__ void tcast_kernel(const float* __restrict__ src, int K, int N,
                             unsigned short* __restrict__ dst, int ldk, int kOff) {
  long long t = blockIdx.x * (long long)blockDim.x + threadIdx.x;
  if (t >= (long long)K * N) return;
  int k = (int)(t / N), n = (int)(t % N);
  dst[(size_t)n * ldk + kOff + k] = f2bf(src[t]);
}

// ------------------------- generic WMMA GEMM -------------------------------
// Y[M x NO] = A[M x K](bf16) @ W[K x NO] + bias, W given as Wt (NO x K bf16).
// act: 0=none, 1=SiLU. 4 waves/block, wave = one 16x16 output tile.
__global__ __launch_bounds__(128) void gemm_bf16_kernel(
    const unsigned short* __restrict__ A, int lda,
    const unsigned short* __restrict__ Wt, int ldb,
    const float* __restrict__ bias,
    float* __restrict__ Y, int ldy,
    int M, int K, int NO, int act) {
  int lane = threadIdx.x & 31;
  int wave = threadIdx.x >> 5;
  int nColTiles = (NO + 15) >> 4;
  int colTile = blockIdx.y * 4 + wave;
  if (colTile >= nColTiles) return;          // wave-uniform exit
  int row0 = blockIdx.x * 16;
  v8f acc = {};
  for (int k = 0; k < K; k += 32) {
    unsigned au[8], bu[8];
    load_fragA(A, lda, row0, M - 1, k, lane, au);
    load_fragB(Wt, ldb, colTile * 16, k, lane, bu);
    acc = wmma_bf16x8(au, bu, acc);
  }
  int col = colTile * 16 + (lane & 15);
  int hi = (lane >> 4) & 1;
  if (col < NO) {
    float bv = bias ? bias[col] : 0.0f;
#pragma unroll
    for (int vg = 0; vg < 8; ++vg) {
      int r = row0 + vg + 8 * hi;
      if (r < M) {
        float v = acc[vg] + bv;
        if (act == 1) v = siluf(v);
        Y[(size_t)r * ldy + col] = v;
      }
    }
  }
}

// ------------------------- edge scatter (segment sum/max/count) ------------
// feat: bf16 [nodes x F]; F/4 lanes per edge, each handles 4 features.
__global__ void scatter_kernel(const unsigned short* __restrict__ feat, int F,
                               const int* __restrict__ src, const int* __restrict__ dst,
                               float* __restrict__ sum, unsigned* __restrict__ mx,
                               float* __restrict__ cnt, int E) {
  int tpe = F >> 2;
  long long t = blockIdx.x * (long long)blockDim.x + threadIdx.x;
  int e = (int)(t / tpe);
  int sub = (int)(t % tpe);
  if (e >= E) return;
  int s = src[e], d = dst[e];
  int f = sub * 4;
  const unsigned* p = (const unsigned*)(feat + (size_t)s * F + f);
  unsigned w0 = p[0], w1 = p[1];
  float v0 = bf2f((unsigned short)(w0 & 0xffff));
  float v1 = bf2f((unsigned short)(w0 >> 16));
  float v2 = bf2f((unsigned short)(w1 & 0xffff));
  float v3 = bf2f((unsigned short)(w1 >> 16));
  float* sp = sum + (size_t)d * F + f;
  unsigned* mp = mx + (size_t)d * F + f;
  atomAddF(sp + 0, v0); atomAddF(sp + 1, v1); atomAddF(sp + 2, v2); atomAddF(sp + 3, v3);
  atomicMax(mp + 0, encOrd(v0)); atomicMax(mp + 1, encOrd(v1));
  atomicMax(mp + 2, encOrd(v2)); atomicMax(mp + 3, encOrd(v3));
  if (sub == 0) atomAddF(cnt + d, 1.0f);
}

// layer-0 finalize: A0 row (32 bf16) = [max(4) | mean(4) | x(4) | 0...]
__global__ void finalize0_kernel(const float* __restrict__ sum, const unsigned* __restrict__ mx,
                                 const float* __restrict__ cnt, const unsigned short* __restrict__ xbf,
                                 unsigned short* __restrict__ A0, int N) {
  int n = blockIdx.x * blockDim.x + threadIdx.x;
  if (n >= N) return;
  float ic = 1.0f / fmaxf(cnt[n], 1.0f);
  unsigned short row[32];
#pragma unroll
  for (int i = 0; i < 4; ++i) {
    float m = decOrd(mx[(size_t)n * 4 + i]);
    if (m == -INFINITY) m = 0.0f;
    row[i] = f2bf(m);
    row[4 + i] = f2bf(sum[(size_t)n * 4 + i] * ic);
    row[8 + i] = xbf[(size_t)n * 4 + i];
  }
#pragma unroll
  for (int i = 12; i < 32; ++i) row[i] = 0;
  uint4* d = (uint4*)(A0 + (size_t)n * 32);
  const uint4* s = (const uint4*)row;
  d[0] = s[0]; d[1] = s[1]; d[2] = s[2]; d[3] = s[3];
}

// hidden-layer finalize: A row (192 bf16) = [max(64) | mean(64) | h(64)]
__global__ void finalizeL_kernel(const float* __restrict__ sum, const unsigned* __restrict__ mx,
                                 const float* __restrict__ cnt, const unsigned short* __restrict__ hbf,
                                 unsigned short* __restrict__ A, int N) {
  long long t = blockIdx.x * (long long)blockDim.x + threadIdx.x;
  int n = (int)(t >> 4), sub = (int)(t & 15);
  if (n >= N) return;
  float ic = 1.0f / fmaxf(cnt[n], 1.0f);
  int f = sub * 4;
  size_t ib = (size_t)n * 64 + f;
  unsigned mpack[2], spack[2];
#pragma unroll
  for (int p2 = 0; p2 < 2; ++p2) {
    float m0 = decOrd(mx[ib + 2 * p2]);     if (m0 == -INFINITY) m0 = 0.0f;
    float m1 = decOrd(mx[ib + 2 * p2 + 1]); if (m1 == -INFINITY) m1 = 0.0f;
    mpack[p2] = pack2(m0, m1);
    spack[p2] = pack2(sum[ib + 2 * p2] * ic, sum[ib + 2 * p2 + 1] * ic);
  }
  unsigned* o0 = (unsigned*)(A + (size_t)n * 192 + f);
  unsigned* o1 = (unsigned*)(A + (size_t)n * 192 + 64 + f);
  unsigned* o2 = (unsigned*)(A + (size_t)n * 192 + 128 + f);
  const unsigned* hp = (const unsigned*)(hbf + ib);
  o0[0] = mpack[0]; o0[1] = mpack[1];
  o1[0] = spack[0]; o1[1] = spack[1];
  o2[0] = hp[0];    o2[1] = hp[1];
}

// ------------------------- LayerNorm (+res, +SiLU) -------------------------
// wave-per-row; F in {64, 256}. outF (f32) and/or outB (bf16) optional.
__global__ void ln_kernel(const float* __restrict__ in, const float* __restrict__ res,
                          const float* __restrict__ g, const float* __restrict__ b,
                          float* __restrict__ outF, unsigned short* __restrict__ outB,
                          int M, int F, int doSilu) {
  int wave = threadIdx.x >> 5;
  int lane = threadIdx.x & 31;
  int row = blockIdx.x * 8 + wave;
  if (row >= M) return;                      // wave-uniform
  const float* rp = in + (size_t)row * F;
  int C = F >> 6;                            // chunks of 64 (2 cols/lane each)
  float vals[8];
  float s = 0.0f;
  for (int i = 0; i < C; ++i) {
    float2 v = *(const float2*)(rp + i * 64 + lane * 2);
    vals[2 * i] = v.x; vals[2 * i + 1] = v.y; s += v.x + v.y;
  }
#pragma unroll
  for (int o = 16; o > 0; o >>= 1) s += __shfl_xor(s, o, 32);
  float mu = s / (float)F;
  float vs = 0.0f;
  for (int i = 0; i < 2 * C; ++i) { vals[i] -= mu; vs += vals[i] * vals[i]; }
#pragma unroll
  for (int o = 16; o > 0; o >>= 1) vs += __shfl_xor(vs, o, 32);
  float inv = rsqrtf(vs / (float)F + 1e-5f);
  for (int i = 0; i < C; ++i) {
    int c0 = i * 64 + lane * 2;
    float y0 = vals[2 * i] * inv * g[c0] + b[c0];
    float y1 = vals[2 * i + 1] * inv * g[c0 + 1] + b[c0 + 1];
    if (res) { y0 += res[(size_t)row * F + c0]; y1 += res[(size_t)row * F + c0 + 1]; }
    if (doSilu) { y0 = siluf(y0); y1 = siluf(y1); }
    if (outF) *(float2*)(outF + (size_t)row * F + c0) = make_float2(y0, y1);
    if (outB) *(unsigned*)(outB + (size_t)row * F + c0) = pack2(y0, y1);
  }
}

// ------------------------- fused node MLP ----------------------------------
// h(64) -> W1(64x256)+b1 -> SiLU -> LN(256) -> W2(256x64)+b2 -> out f32
// block = 512 threads = 16 waves, 16 nodes per block.
__global__ __launch_bounds__(512) void mlp_fused_kernel(
    const unsigned short* __restrict__ hbf,   // N x 64 bf16
    const unsigned short* __restrict__ W1t,   // 256 x 64 bf16 (N x K)
    const float* __restrict__ b1,
    const float* __restrict__ g, const float* __restrict__ bn,
    const unsigned short* __restrict__ W2t,   // 64 x 256 bf16 (N x K)
    const float* __restrict__ b2,
    float* __restrict__ out, int M) {
  __shared__ unsigned sA[16 * 32];            // 16 rows x 64 bf16
  __shared__ float    sHid[16][264];          // hidden f32 (+pad)
  __shared__ unsigned sNrm[16][132];          // normalized bf16 packed (+pad)
  int lane = threadIdx.x & 31;
  int wave = threadIdx.x >> 5;                // 0..15
  int row0 = blockIdx.x * 16;
  {
    int tt = threadIdx.x;                     // stage A tile
    int r = tt >> 5, c = tt & 31;
    int rr = row0 + r; if (rr > M - 1) rr = M - 1;
    sA[tt] = *(const unsigned*)(hbf + (size_t)rr * 64 + c * 2);
  }
  __syncthreads();
  // GEMM1: wave = hidden col tile (16 tiles x 16 = 256), K=64
  {
    v8f acc = {};
    int r = lane & 15, hi = (lane >> 4) & 1;
    for (int k = 0; k < 64; k += 32) {
      unsigned au[8], bu[8];
#pragma unroll
      for (int j = 0; j < 8; ++j) {
        int k0 = 2 * j + 8 * hi + ((j >= 4) ? 8 : 0) + k;
        au[j] = sA[r * 32 + (k0 >> 1)];
      }
      load_fragB(W1t, 64, wave * 16, k, lane, bu);
      acc = wmma_bf16x8(au, bu, acc);
    }
    int c = wave * 16 + (lane & 15);
    float bv = b1[c];
#pragma unroll
    for (int vg = 0; vg < 8; ++vg) sHid[vg + 8 * hi][c] = siluf(acc[vg] + bv);
  }
  __syncthreads();
  // LN per row (wave w -> row w), write normalized bf16 to sNrm
  {
    int r = wave;
    float vals[8];
    float s = 0.0f;
#pragma unroll
    for (int i = 0; i < 4; ++i) {
      int c0 = i * 64 + lane * 2;
      vals[2 * i] = sHid[r][c0]; vals[2 * i + 1] = sHid[r][c0 + 1];
      s += vals[2 * i] + vals[2 * i + 1];
    }
#pragma unroll
    for (int o = 16; o > 0; o >>= 1) s += __shfl_xor(s, o, 32);
    float mu = s * (1.0f / 256.0f);
    float vs = 0.0f;
#pragma unroll
    for (int i = 0; i < 8; ++i) { vals[i] -= mu; vs += vals[i] * vals[i]; }
#pragma unroll
    for (int o = 16; o > 0; o >>= 1) vs += __shfl_xor(vs, o, 32);
    float inv = rsqrtf(vs * (1.0f / 256.0f) + 1e-5f);
#pragma unroll
    for (int i = 0; i < 4; ++i) {
      int c0 = i * 64 + lane * 2;
      float y0 = vals[2 * i] * inv * g[c0] + bn[c0];
      float y1 = vals[2 * i + 1] * inv * g[c0 + 1] + bn[c0 + 1];
      sNrm[r][i * 32 + lane] = pack2(y0, y1);
    }
  }
  __syncthreads();
  // GEMM2: waves 0..3, col tile = wave (64 out), K=256
  if (wave < 4) {
    v8f acc = {};
    int r = lane & 15, hi = (lane >> 4) & 1;
    for (int k = 0; k < 256; k += 32) {
      unsigned au[8], bu[8];
#pragma unroll
      for (int j = 0; j < 8; ++j) {
        int k0 = 2 * j + 8 * hi + ((j >= 4) ? 8 : 0) + k;
        au[j] = sNrm[r][k0 >> 1];
      }
      load_fragB(W2t, 256, wave * 16, k, lane, bu);
      acc = wmma_bf16x8(au, bu, acc);
    }
    int c = wave * 16 + (lane & 15);
    float bv = b2[c];
#pragma unroll
    for (int vg = 0; vg < 8; ++vg) {
      int rr = row0 + vg + 8 * hi;
      if (rr < M) out[(size_t)rr * 64 + c] = acc[vg] + bv;
    }
  }
}

// ------------------------- graph pooling -----------------------------------
__global__ void pool_kernel(const float* __restrict__ h, const int* __restrict__ batch,
                            float* __restrict__ gsum, unsigned* __restrict__ gmax,
                            float* __restrict__ gcnt, int N) {
  long long t = blockIdx.x * (long long)blockDim.x + threadIdx.x;
  int n = (int)(t >> 4), sub = (int)(t & 15);
  if (n >= N) return;
  int gidx = batch[n];
  int f = sub * 4;
  float4 v = *(const float4*)(h + (size_t)n * 64 + f);
  float* sp = gsum + (size_t)gidx * 64 + f;
  unsigned* mp = gmax + (size_t)gidx * 64 + f;
  atomAddF(sp + 0, v.x); atomAddF(sp + 1, v.y); atomAddF(sp + 2, v.z); atomAddF(sp + 3, v.w);
  atomicMax(mp + 0, encOrd(v.x)); atomicMax(mp + 1, encOrd(v.y));
  atomicMax(mp + 2, encOrd(v.z)); atomicMax(mp + 3, encOrd(v.w));
  if (sub == 0) atomAddF(gcnt + gidx, 1.0f);
}

// o row (224 bf16) = [mean(64) | max(64) | sum(64) | root_x(4) | 0 pad(28)]
__global__ void build_o_kernel(const float* __restrict__ gsum, const unsigned* __restrict__ gmax,
                               const float* __restrict__ gcnt, const float* __restrict__ x,
                               const int* __restrict__ ptrArr, unsigned short* __restrict__ o,
                               int G) {
  int gidx = blockIdx.x;
  int c = threadIdx.x;
  if (gidx >= G || c >= 224) return;
  float v = 0.0f;
  if (c < 64) {
    v = gsum[(size_t)gidx * 64 + c] / fmaxf(gcnt[gidx], 1.0f);
  } else if (c < 128) {
    float m = decOrd(gmax[(size_t)gidx * 64 + (c - 64)]);
    v = (m == -INFINITY) ? 0.0f : m;
  } else if (c < 192) {
    v = gsum[(size_t)gidx * 64 + (c - 128)];
  } else if (c < 196) {
    v = x[(size_t)ptrArr[gidx] * 4 + (c - 192)];
  }
  o[(size_t)gidx * 224 + c] = f2bf(v);
}

// ---------------------------------------------------------------------------
extern "C" void kernel_launch(void* const* d_in, const int* in_sizes, int n_in,
                              void* d_out, int out_size, void* d_ws, size_t ws_size,
                              hipStream_t stream) {
  const int N = in_sizes[0] / 4;
  const int E = in_sizes[1] / 2;
  const int G = in_sizes[3] - 1;

  const float* x    = (const float*)d_in[0];
  const int* ei     = (const int*)d_in[1];
  const int* srcIdx = ei;
  const int* dstIdx = ei + E;
  const int* batch  = (const int*)d_in[2];
  const int* ptrArr = (const int*)d_in[3];
  const float* Wl0  = (const float*)d_in[4];
  const float* bl0  = (const float*)d_in[5];
  const float* Wr0  = (const float*)d_in[6];
  const float* Wl   = (const float*)d_in[7];
  const float* bl   = (const float*)d_in[8];
  const float* Wr   = (const float*)d_in[9];
  const float* Wres = (const float*)d_in[10];
  const float* bres = (const float*)d_in[11];
  const float* ln_g = (const float*)d_in[12];
  const float* ln_b = (const float*)d_in[13];
  const float* mW1  = (const float*)d_in[14];
  const float* mb1  = (const float*)d_in[15];
  const float* mg   = (const float*)d_in[16];
  const float* mbn  = (const float*)d_in[17];
  const float* mW2  = (const float*)d_in[18];
  const float* mb2  = (const float*)d_in[19];
  const float* rW1  = (const float*)d_in[20];
  const float* rb1  = (const float*)d_in[21];
  const float* rg   = (const float*)d_in[22];
  const float* rbn  = (const float*)d_in[23];
  const float* rW2  = (const float*)d_in[24];
  const float* rb2  = (const float*)d_in[25];

  char* base = (char*)d_ws;
  size_t off = 0;
  auto alloc = [&](size_t bytes) -> char* {
    off = (off + 255) & ~(size_t)255;
    char* p = base + off;
    off += bytes;
    return p;
  };
  float* hA            = (float*)alloc((size_t)N * 64 * 4);
  float* hB            = (float*)alloc((size_t)N * 64 * 4);
  float* hpre          = (float*)alloc((size_t)N * 64 * 4);   // also MLP output
  unsigned short* hbf  = (unsigned short*)alloc((size_t)N * 64 * 2);
  unsigned short* xbf  = (unsigned short*)alloc((size_t)N * 4 * 2);
  float* asum          = (float*)alloc((size_t)N * 64 * 4);
  unsigned* amax       = (unsigned*)alloc((size_t)N * 64 * 4);
  float* acnt          = (float*)alloc((size_t)N * 4);
  unsigned short* Abuf = (unsigned short*)alloc((size_t)N * 192 * 2);
  float* gsum          = (float*)alloc((size_t)G * 64 * 4);
  unsigned* gmax       = (unsigned*)alloc((size_t)G * 64 * 4);
  float* gcnt          = (float*)alloc((size_t)G * 4);
  unsigned short* obf  = (unsigned short*)alloc((size_t)G * 224 * 2);
  float* rohid         = (float*)alloc((size_t)G * 256 * 4);
  unsigned short* o2bf = (unsigned short*)alloc((size_t)G * 256 * 2);
  size_t wStart = (off + 255) & ~(size_t)255;
  unsigned short* W0t   = (unsigned short*)alloc(64 * 32 * 2);
  unsigned short* Wrest = (unsigned short*)alloc(64 * 32 * 2);
  unsigned short* Wlt   = (unsigned short*)alloc(3 * 64 * 192 * 2);
  unsigned short* W1t   = (unsigned short*)alloc(256 * 64 * 2);
  unsigned short* W2t   = (unsigned short*)alloc(64 * 256 * 2);
  unsigned short* rW1t  = (unsigned short*)alloc(256 * 224 * 2);
  unsigned short* rW2t  = (unsigned short*)alloc(16 * 256 * 2);
  size_t wBytes = off - wStart;

  auto nb = [](long long n, int b) { return dim3((unsigned)((n + b - 1) / b)); };
  auto gemm = [&](const unsigned short* A, int lda, const unsigned short* Wt, int ldb,
                  const float* bias, float* Y, int ldy, int M, int K, int NO, int act) {
    int nct = (NO + 15) / 16;
    dim3 grid((unsigned)((M + 15) / 16), (unsigned)((nct + 3) / 4));
    gemm_bf16_kernel<<<grid, 128, 0, stream>>>(A, lda, Wt, ldb, bias, Y, ldy, M, K, NO, act);
  };

  // ---- weight prep: zero padded region, then transpose+cast to bf16 (NxK)
  fill_u32_kernel<<<nb((long long)(wBytes / 4), 256), 256, 0, stream>>>(
      (unsigned*)(base + wStart), 0u, (long long)(wBytes / 4));
  tcast_kernel<<<nb(8 * 64, 256), 256, 0, stream>>>(Wl0, 8, 64, W0t, 32, 0);
  tcast_kernel<<<nb(4 * 64, 256), 256, 0, stream>>>(Wr0, 4, 64, W0t, 32, 8);
  tcast_kernel<<<nb(4 * 64, 256), 256, 0, stream>>>(Wres, 4, 64, Wrest, 32, 8);  // aligns with x @ cols 8-11
  for (int i = 0; i < 3; ++i) {
    tcast_kernel<<<nb(128 * 64, 256), 256, 0, stream>>>(Wl + (size_t)i * 128 * 64, 128, 64,
                                                        Wlt + (size_t)i * 64 * 192, 192, 0);
    tcast_kernel<<<nb(64 * 64, 256), 256, 0, stream>>>(Wr + (size_t)i * 64 * 64, 64, 64,
                                                       Wlt + (size_t)i * 64 * 192, 192, 128);
  }
  tcast_kernel<<<nb(64 * 256, 256), 256, 0, stream>>>(mW1, 64, 256, W1t, 64, 0);
  tcast_kernel<<<nb(256 * 64, 256), 256, 0, stream>>>(mW2, 256, 64, W2t, 256, 0);
  tcast_kernel<<<nb(196 * 256, 256), 256, 0, stream>>>(rW1, 196, 256, rW1t, 224, 0);
  tcast_kernel<<<nb(256 * 4, 256), 256, 0, stream>>>(rW2, 256, 4, rW2t, 256, 0);
  cast_f32_bf16_kernel<<<nb((long long)N * 4, 256), 256, 0, stream>>>(x, xbf, (long long)N * 4);

  // ---- layer 0: aggregate x (F=4), fused [max|mean|x] @ [Wl0;Wr0], x_res
  fill_f32_kernel<<<nb((long long)N * 4, 256), 256, 0, stream>>>(asum, 0.0f, (long long)N * 4);
  fill_u32_kernel<<<nb((long long)N * 4, 256), 256, 0, stream>>>(amax, ENC_NEGINF, (long long)N * 4);
  fill_f32_kernel<<<nb(N, 256), 256, 0, stream>>>(acnt, 0.0f, N);
  scatter_kernel<<<nb((long long)E, 256), 256, 0, stream>>>(xbf, 4, srcIdx, dstIdx, asum, amax, acnt, E);
  finalize0_kernel<<<nb(N, 256), 256, 0, stream>>>(asum, amax, acnt, xbf, Abuf, N);
  gemm(Abuf, 32, W0t, 32, bl0, hpre, 64, N, 32, 64, 0);    // sage pre-act
  gemm(Abuf, 32, Wrest, 32, bres, hB, 64, N, 32, 64, 0);   // x_res (uses x cols 8-11)
  ln_kernel<<<nb((long long)(N + 7) / 8, 1), 256, 0, stream>>>(hpre, hB, ln_g, ln_b, hA, hbf, N, 64, 1);

  // ---- layers 1..3
  float* cur = hA;
  float* oth = hB;
  for (int i = 0; i < 3; ++i) {
    fill_f32_kernel<<<nb((long long)N * 64, 256), 256, 0, stream>>>(asum, 0.0f, (long long)N * 64);
    fill_u32_kernel<<<nb((long long)N * 64, 256), 256, 0, stream>>>(amax, ENC_NEGINF, (long long)N * 64);
    fill_f32_kernel<<<nb(N, 256), 256, 0, stream>>>(acnt, 0.0f, N);
    scatter_kernel<<<nb((long long)E * 16, 256), 256, 0, stream>>>(hbf, 64, srcIdx, dstIdx, asum, amax, acnt, E);
    finalizeL_kernel<<<nb((long long)N * 16, 256), 256, 0, stream>>>(asum, amax, acnt, hbf, Abuf, N);
    gemm(Abuf, 192, Wlt + (size_t)i * 64 * 192, 192, bl + (size_t)i * 64, hpre, 64, N, 192, 64, 0);
    ln_kernel<<<nb((long long)(N + 7) / 8, 1), 256, 0, stream>>>(
        hpre, cur, ln_g + (size_t)(i + 1) * 64, ln_b + (size_t)(i + 1) * 64, oth, hbf, N, 64, 1);
    float* t2 = cur; cur = oth; oth = t2;
  }

  // ---- fused node MLP (h -> 256 -> LN -> 64), output f32 into hpre
  mlp_fused_kernel<<<nb((long long)(N + 15) / 16, 1), 512, 0, stream>>>(
      hbf, W1t, mb1, mg, mbn, W2t, mb2, hpre, N);

  // ---- graph pooling
  fill_f32_kernel<<<nb((long long)G * 64, 256), 256, 0, stream>>>(gsum, 0.0f, (long long)G * 64);
  fill_u32_kernel<<<nb((long long)G * 64, 256), 256, 0, stream>>>(gmax, ENC_NEGINF, (long long)G * 64);
  fill_f32_kernel<<<nb(G, 256), 256, 0, stream>>>(gcnt, 0.0f, G);
  pool_kernel<<<nb((long long)N * 16, 256), 256, 0, stream>>>(hpre, batch, gsum, gmax, gcnt, N);
  build_o_kernel<<<dim3((unsigned)G), 224, 0, stream>>>(gsum, gmax, gcnt, x, ptrArr, obf, G);

  // ---- readout: o(224pad) @ roW1 + b1 -> SiLU -> LN -> @ roW2 + b2
  gemm(obf, 224, rW1t, 224, rb1, rohid, 256, G, 224, 256, 1);
  ln_kernel<<<nb((long long)(G + 7) / 8, 1), 256, 0, stream>>>(
      rohid, nullptr, rg, rbn, nullptr, o2bf, G, 256, 0);
  gemm(o2bf, 256, rW2t, 256, rb2, (float*)d_out, 4, G, 256, 4, 0);
}
